// CausalSelfAttention_549755814167
// MI455X (gfx1250) — compile-verified, matrix operand order checked by reading, not run on hardware
//
#include <hip/hip_runtime.h>
#include <cstdint>

// ---------------------------------------------------------------------------
// CDNA5 (gfx1250) causal self-attention, bf16 WMMA path, fp32 accumulate.
// 64x32 output tile per wave: 8 v_wmma_f32_16x16x32_bf16 per K=32 step vs
// 12 global_load_b128. Explicit address_space(1) pointers keep the memory
// path on GLOBAL (LOADcnt only) instead of FLAT (LOADcnt+DScnt).
// ---------------------------------------------------------------------------

typedef unsigned short u16;
typedef __attribute__((ext_vector_type(16))) __bf16        v16bf;
typedef __attribute__((ext_vector_type(8)))  float         v8f;
typedef __attribute__((ext_vector_type(4)))  unsigned int  v4u;  // POD 128-bit

#define AS1 __attribute__((address_space(1)))
typedef const AS1 u16*  gcu16;
typedef AS1 u16*        gu16;
typedef AS1 float*      gf32;
typedef const AS1 v4u*  gcv4;

union FragU {
  struct { v4u lo, hi; } u;
  v16bf v;
};

// fp32 -> bf16 round-to-nearest-even
__device__ __forceinline__ u16 f2bf(float f) {
  union { float f; unsigned int u; } c;
  c.f = f;
  unsigned int u = c.u;
  u += 0x7FFFu + ((u >> 16) & 1u);
  return (u16)(u >> 16);
}

// ---------------------------------------------------------------------------
// Fragment load from a precomputed per-lane GLOBAL pointer.
// 16-bit A-matrix 16x32 layout (ISA 7.12.2): lane L holds row M = L%16;
// half = L/16 selects the K group:
//   VGPR0..3 : K = half*8 + 0..7      (16 contiguous bytes)
//   VGPR4..7 : K = 16 + half*8 + 0..7 (next 16 contiguous bytes, +32B away)
// B mirrors this with lane -> column N, so row-major [N x K] operands
// (W[o,d], K[k,d], V^T[d,k]) use the same loader.
// ---------------------------------------------------------------------------
__device__ __forceinline__ v16bf ld_frag(gcu16 p) {
  FragU f;
  f.u.lo = *(gcv4)p;         // K offsets 0..7 of half
  f.u.hi = *(gcv4)(p + 16);  // K offsets 16..23 of half
  return f.v;
}

// 32-bit C/D 16x16 layout: VGPR i, lane L -> M = i + (L/16)*8, N = L%16.
__device__ __forceinline__ void store_tile_bf16(gu16 out, int ld,
                                                int m0, int n0, v8f acc) {
  const int lane = threadIdx.x & 31;
  const int n  = n0 + (lane & 15);
  const int mb = m0 + ((lane >> 4) << 3);
#pragma unroll
  for (int i = 0; i < 8; ++i)
    out[(size_t)(mb + i) * ld + n] = f2bf(acc[i]);
}

__device__ __forceinline__ void store_tile_f32(gf32 out, int ld,
                                               int m0, int n0, v8f acc) {
  const int lane = threadIdx.x & 31;
  const int n  = n0 + (lane & 15);
  const int mb = m0 + ((lane >> 4) << 3);
#pragma unroll
  for (int i = 0; i < 8; ++i)
    out[(size_t)(mb + i) * ld + n] = acc[i];
}

// masked + scaled store for attention scores: entry (q=m, k=n) gets -inf when n>m
__device__ __forceinline__ void store_tile_scores(gf32 out, int ld,
                                                  int m0, int n0, v8f acc,
                                                  float scale) {
  const int lane = threadIdx.x & 31;
  const int n  = n0 + (lane & 15);
  const int mb = m0 + ((lane >> 4) << 3);
#pragma unroll
  for (int i = 0; i < 8; ++i) {
    const int m = mb + i;
    out[(size_t)m * ld + n] = (n > m) ? -__builtin_inff() : acc[i] * scale;
  }
}

#define WMMA_BF16(A, B, C) \
  __builtin_amdgcn_wmma_f32_16x16x32_bf16(false, (A), false, (B), (short)0, (C), false, false)

// ---------------------------------------------------------------------------
// fp32 -> bf16 conversion (grid-stride)
// ---------------------------------------------------------------------------
__global__ void cvt_f32_bf16(const float* __restrict__ src, u16* __restrict__ dst,
                             size_t n) {
  size_t i = (size_t)blockIdx.x * blockDim.x + threadIdx.x;
  const size_t stride = (size_t)gridDim.x * blockDim.x;
  for (; i < n; i += stride) dst[i] = f2bf(src[i]);
}

// ---------------------------------------------------------------------------
// 64x32 wave-tile GEMM body with loop-carried per-lane GLOBAL pointers.
// C[m,n] = sum_k A[m,k]*B[n,k]; A: MxK row-major bf16, B: NxK row-major bf16.
// ---------------------------------------------------------------------------
struct Acc64x32 {
  v8f c[4][2];
};

__device__ __forceinline__ void gemm_tile_64x32(gcu16 A, gcu16 B,
                                                int m0, int n0, int ldA, int ldB,
                                                int kEnd, Acc64x32& acc) {
  const int lane = threadIdx.x & 31;
  const int r = lane & 15;
  const int h = (lane >> 4) << 3;

  gcu16 pA[4];
  gcu16 pB[2];
#pragma unroll
  for (int i = 0; i < 4; ++i) pA[i] = A + (size_t)(m0 + 16 * i + r) * ldA + h;
#pragma unroll
  for (int j = 0; j < 2; ++j) pB[j] = B + (size_t)(n0 + 16 * j + r) * ldB + h;

  for (int k0 = 0; k0 < kEnd; k0 += 32) {
    v16bf a[4], b[2];
#pragma unroll
    for (int i = 0; i < 4; ++i) a[i] = ld_frag(pA[i]);
#pragma unroll
    for (int j = 0; j < 2; ++j) b[j] = ld_frag(pB[j]);
    // Unconditional speculative prefetch of the next K-step (dropped silently
    // if past the allocation; keeps the CFG branch-free).
    __builtin_prefetch(pA[0] + 32, 0, 3);
    __builtin_prefetch(pB[0] + 32, 0, 3);
#pragma unroll
    for (int i = 0; i < 4; ++i)
#pragma unroll
      for (int j = 0; j < 2; ++j)
        acc.c[i][j] = WMMA_BF16(a[i], b[j], acc.c[i][j]);
#pragma unroll
    for (int i = 0; i < 4; ++i) pA[i] += 32;
#pragma unroll
    for (int j = 0; j < 2; ++j) pB[j] += 32;
  }
}

// ---------------------------------------------------------------------------
// QKV projection GEMM, bf16 output.
// ---------------------------------------------------------------------------
__global__ void __launch_bounds__(256)
gemm_bf16_kernel(const u16* __restrict__ Ag, const u16* __restrict__ Bg,
                 u16* __restrict__ Cg, int M, int Nn, int K) {
  gcu16 A = (gcu16)Ag;
  gcu16 B = (gcu16)Bg;
  gu16  C = (gu16)Cg;

  const int wave   = threadIdx.x >> 5;
  const int wid    = blockIdx.x * (blockDim.x >> 5) + wave;
  const int ntiles = Nn >> 5;                 // 32-wide N tiles
  const int total  = (M >> 6) * ntiles;       // 64-tall M tiles
  if (wid >= total) return;                   // wave-uniform: EXEC all-ones
  const int m0 = (wid / ntiles) << 6;
  const int n0 = (wid % ntiles) << 5;

  Acc64x32 acc = {};
  gemm_tile_64x32(A, B, m0, n0, K, K, K, acc);

#pragma unroll
  for (int i = 0; i < 4; ++i)
#pragma unroll
    for (int j = 0; j < 2; ++j)
      store_tile_bf16(C, Nn, m0 + 16 * i, n0 + 16 * j, acc.c[i][j]);
}

// ---------------------------------------------------------------------------
// scores[b,q,k] = scale * sum_d Q[b,q,d]*K[b,k,d], causal mask fused (-inf k>q)
// ---------------------------------------------------------------------------
__global__ void __launch_bounds__(256)
scores_kernel(const u16* __restrict__ Qg, const u16* __restrict__ Kg,
              float* __restrict__ Sg, int N, int D, float scale) {
  const int b = blockIdx.z;
  gcu16 Qb = (gcu16)Qg + (size_t)b * N * D;
  gcu16 Kb = (gcu16)Kg + (size_t)b * N * D;
  gf32  Sb = (gf32)Sg  + (size_t)b * N * N;

  const int wave   = threadIdx.x >> 5;
  const int wid    = blockIdx.x * (blockDim.x >> 5) + wave;
  const int ntiles = N >> 5;                  // key tiles (32 wide)
  const int total  = (N >> 6) * ntiles;       // query tiles (64 tall)
  if (wid >= total) return;
  const int m0 = (wid / ntiles) << 6;         // query tile base
  const int n0 = (wid % ntiles) << 5;         // key tile base
  const bool live = (n0 <= m0 + 63);          // any unmasked element?

  Acc64x32 acc = {};
  if (live) gemm_tile_64x32(Qb, Kb, m0, n0, D, D, D, acc);
  const float sc = live ? scale : 0.f;
#pragma unroll
  for (int i = 0; i < 4; ++i)
#pragma unroll
    for (int j = 0; j < 2; ++j)
      store_tile_scores(Sb, N, m0 + 16 * i, n0 + 16 * j, acc.c[i][j], sc);
}

// ---------------------------------------------------------------------------
// Softmax over the QUERY axis (reference uses axis=1): per (b, key-column k),
// normalize over q in [k, N). One thread per column; adjacent threads touch
// adjacent addresses -> coalesced. Output attn as bf16.
// ---------------------------------------------------------------------------
__global__ void softmax_q_kernel(const float* __restrict__ Sg, u16* __restrict__ Pg,
                                 int N) {
  const int b = blockIdx.y;
  const int k = blockIdx.x * blockDim.x + threadIdx.x;
  if (k >= N) return;
  const AS1 float* s = (const AS1 float*)Sg + (size_t)b * N * N;
  gu16             p = (gu16)Pg + (size_t)b * N * N;

  float m = -__builtin_inff();
  for (int q = k; q < N; ++q) m = fmaxf(m, s[(size_t)q * N + k]);
  float sum = 0.f;
  for (int q = k; q < N; ++q) sum += __expf(s[(size_t)q * N + k] - m);
  const float inv = 1.0f / sum;
  for (int q = 0; q < N; ++q) {
    const float v = (q < k) ? 0.f : __expf(s[(size_t)q * N + k] - m) * inv;
    p[(size_t)q * N + k] = f2bf(v);
  }
}

// ---------------------------------------------------------------------------
// bf16 transpose: V[b, N, D] -> Vt[b, D, N] (LDS-tiled, padded vs conflicts)
// ---------------------------------------------------------------------------
__global__ void transpose_bf16_kernel(const u16* __restrict__ Vg, u16* __restrict__ Vtg,
                                      int N, int D) {
  __shared__ u16 tile[32][33];
  const int b  = blockIdx.z;
  const int d0 = blockIdx.x * 32;
  const int n0 = blockIdx.y * 32;
  gcu16 src = (gcu16)Vg  + (size_t)b * N * D;
  gu16  dst = (gu16)Vtg + (size_t)b * N * D;   // D*N elements per batch
  const int tx = threadIdx.x & 31;
  const int ty = threadIdx.x >> 5;             // 0..7
  for (int i = ty; i < 32; i += 8)
    tile[i][tx] = src[(size_t)(n0 + i) * D + d0 + tx];
  __syncthreads();
  for (int i = ty; i < 32; i += 8)
    dst[(size_t)(d0 + i) * N + n0 + tx] = tile[tx][i];
}

// ---------------------------------------------------------------------------
// out[b,q,d] = sum_k attn[b,q,k] * V[b,k,d] = sum_k P[q,k] * Vt[d,k]
// K-loop truncated at the causal boundary (attn[q,k]==0 for k>q). fp32 out.
// ---------------------------------------------------------------------------
__global__ void __launch_bounds__(256)
av_kernel(const u16* __restrict__ Pg, const u16* __restrict__ Vtg,
          float* __restrict__ Og, int N, int D) {
  const int b = blockIdx.z;
  gcu16 Pb = (gcu16)Pg  + (size_t)b * N * N;
  gcu16 Vb = (gcu16)Vtg + (size_t)b * (size_t)D * N;
  gf32  Ob = (gf32)Og   + (size_t)b * N * D;

  const int wave   = threadIdx.x >> 5;
  const int wid    = blockIdx.x * (blockDim.x >> 5) + wave;
  const int ntiles = D >> 5;                  // feature tiles (32 wide)
  const int total  = (N >> 6) * ntiles;       // query tiles (64 tall)
  if (wid >= total) return;
  const int m0 = (wid / ntiles) << 6;         // query tile base
  const int n0 = (wid % ntiles) << 5;         // feature tile base
  const int kEnd = m0 + 64;                   // causal truncation (tile-aligned)

  Acc64x32 acc = {};
  gemm_tile_64x32(Pb, Vb, m0, n0, N, N, kEnd, acc);

#pragma unroll
  for (int i = 0; i < 4; ++i)
#pragma unroll
    for (int j = 0; j < 2; ++j)
      store_tile_f32(Ob, D, m0 + 16 * i, n0 + 16 * j, acc.c[i][j]);
}

// ---------------------------------------------------------------------------
// Host-side orchestration (all on `stream`, graph-capture safe)
// ---------------------------------------------------------------------------
extern "C" void kernel_launch(void* const* d_in, const int* in_sizes, int n_in,
                              void* d_out, int out_size, void* d_ws, size_t ws_size,
                              hipStream_t stream) {
  const int B = 4, N = 2048, D = 1024;
  const int M = B * N;                 // 8192 rows of x
  const float scale = 1.0f / 32.0f;    // 1/sqrt(D)

  const float* x  = (const float*)d_in[0];
  const float* Wq = (const float*)d_in[1];
  const float* Wk = (const float*)d_in[2];
  const float* Wv = (const float*)d_in[3];
  float* out = (float*)d_out;

  // workspace layout (flat, 256B-aligned chunks)
  char* ws = (char*)d_ws;
  auto take = [&](size_t bytes) {
    char* p = ws;
    ws += (bytes + 255) & ~(size_t)255;
    return p;
  };
  u16*   xb  = (u16*)take((size_t)M * D * 2);        // 16 MB
  u16*   wqb = (u16*)take((size_t)D * D * 2);        //  2 MB
  u16*   wkb = (u16*)take((size_t)D * D * 2);
  u16*   wvb = (u16*)take((size_t)D * D * 2);
  u16*   Qb  = (u16*)take((size_t)M * D * 2);        // 16 MB
  u16*   Kb  = (u16*)take((size_t)M * D * 2);
  u16*   Vb  = (u16*)take((size_t)M * D * 2);
  u16*   Vt  = (u16*)take((size_t)M * D * 2);
  float* S   = (float*)take((size_t)B * N * N * 4);  // 64 MB
  u16*   P   = (u16*)take((size_t)B * N * N * 2);    // 32 MB

  // 1) fp32 -> bf16
  cvt_f32_bf16<<<1024, 256, 0, stream>>>(x,  xb,  (size_t)M * D);
  cvt_f32_bf16<<<256,  256, 0, stream>>>(Wq, wqb, (size_t)D * D);
  cvt_f32_bf16<<<256,  256, 0, stream>>>(Wk, wkb, (size_t)D * D);
  cvt_f32_bf16<<<256,  256, 0, stream>>>(Wv, wvb, (size_t)D * D);

  // 2) Q/K/V projections: 8192x1024x1024, 64x32 tile per wave, 8 waves/block
  {
    const int tiles  = (M / 64) * (D / 32);    // 4096
    const int blocks = (tiles + 7) / 8;        // 512
    gemm_bf16_kernel<<<blocks, 256, 0, stream>>>(xb, wqb, Qb, M, D, D);
    gemm_bf16_kernel<<<blocks, 256, 0, stream>>>(xb, wkb, Kb, M, D, D);
    gemm_bf16_kernel<<<blocks, 256, 0, stream>>>(xb, wvb, Vb, M, D, D);
  }

  // 3) V -> V^T (per batch) for row-contiguous B-fragments in the AV GEMM
  transpose_bf16_kernel<<<dim3(D / 32, N / 32, B), 256, 0, stream>>>(Vb, Vt, N, D);

  // 4) scores = scale * Q K^T with fused causal mask
  {
    const int tiles  = (N / 64) * (N / 32);    // 2048 per batch
    const int blocks = (tiles + 7) / 8;        // 256
    scores_kernel<<<dim3(blocks, 1, B), 256, 0, stream>>>(Qb, Kb, S, N, D, scale);
  }

  // 5) softmax over query axis (axis=1), write bf16 attn
  softmax_q_kernel<<<dim3(N / 256, B), 256, 0, stream>>>(S, P, N);

  // 6) out = attn @ V (causally truncated K-loop)
  {
    const int tiles  = (N / 64) * (D / 32);    // 1024 per batch
    const int blocks = (tiles + 7) / 8;        // 128
    av_kernel<<<dim3(blocks, 1, B), 256, 0, stream>>>(P, Vt, out, N, D);
  }
}
